// Emformer_36816459661587
// MI455X (gfx1250) — compile-verified, matrix operand order checked by reading, not run on hardware
//
#include <hip/hip_runtime.h>
#include <hip/hip_bf16.h>
#include <math.h>

// ---------------- types for WMMA ----------------
typedef _Float16 half8 __attribute__((ext_vector_type(8)));
typedef _Float16 v16h  __attribute__((ext_vector_type(16)));
typedef float    v8f   __attribute__((ext_vector_type(8)));

// ---------------- problem constants ----------------
#define B_    4
#define TIN   1028
#define FEAT_ 80
#define D_    512
#define H_    8
#define L_    20
#define FF_   2048
#define SEG_  4
#define LC_   30
#define OUTW  1024
#define T_    256          // time after 2x conv subsample
#define BT    1024         // B_*T_
#define T1    513          // conv1 out time
#define F1    39           // conv1 out freq
#define F2C   19           // conv2 out freq
#define K2    4608         // 512*9 (implicit im2col K)
#define M2    19456        // B_*T_*F2C
#define SLK   9728         // D_*F2C

// ============================================================
// f32 -> f16 conversion (weights, grid-stride)
// ============================================================
__global__ void f32_to_f16_kernel(const float* __restrict__ in,
                                  _Float16* __restrict__ out, long n) {
  long i = (long)blockIdx.x * blockDim.x + threadIdx.x;
  long stride = (long)gridDim.x * blockDim.x;
  for (; i < n; i += stride) out[i] = (_Float16)in[i];
}

// ============================================================
// conv1: (B,1,1028,80) -> relu -> f16 (B,512,513,39)
// ============================================================
__global__ void conv1_kernel(const float* __restrict__ feats,
                             const float* __restrict__ w,
                             const float* __restrict__ bias,
                             _Float16* __restrict__ out) {
  long idx = (long)blockIdx.x * 256 + threadIdx.x;
  const long total = (long)B_ * D_ * T1 * F1;
  if (idx >= total) return;
  int xx = (int)(idx % F1); long r = idx / F1;
  int y  = (int)(r % T1);   r /= T1;
  int co = (int)(r % D_);   int b = (int)(r / D_);
  const float* wp = w + (long)co * 9;
  const float* fp = feats + ((long)b * TIN + 2 * y) * FEAT_ + 2 * xx;
  float acc = bias[co];
#pragma unroll
  for (int kh = 0; kh < 3; ++kh)
#pragma unroll
    for (int kw = 0; kw < 3; ++kw)
      acc += fp[kh * FEAT_ + kw] * wp[kh * 3 + kw];
  out[idx] = (_Float16)fmaxf(acc, 0.f);
}

// ============================================================
// Generic WMMA GEMM: C[M,N] = A[M,K] * W[N,K]^T (+bias)(+resid)(act)
// A,W f16 row-major (K contiguous). 256 thr, block tile 128x128, K-step 64.
// Double-buffered staging via GLOBAL_LOAD_ASYNC_TO_LDS_B128 (ASYNCcnt),
// overlapping HBM->LDS of stage k+1 with the WMMA core on stage k.
// The smem pointer is passed into the asm as a dummy operand so the shared
// array address-escapes: without it, alias analysis proves smem is never
// written and folds the ds_load fragment reads to undef.
// act: 0 none, 1 relu, 2 exact gelu
// ============================================================
__global__ __launch_bounds__(256)
void gemm_wmma(const _Float16* __restrict__ A, const _Float16* __restrict__ W,
               const float* __restrict__ bias, const float* __restrict__ resid,
               float* __restrict__ outF, _Float16* __restrict__ outH,
               int M, int N, int K, int act) {
  // Single shared block => LDS byte offset 0 for async-to-LDS addressing.
  // Layout: stage buf at byte buf*64KB: A(16KB) at +0, W(16KB) at +32KB.
  __shared__ __align__(16) _Float16 smem[2 * 2 * 128 * 64];
  const int tid  = threadIdx.x;
  const int lane = tid & 31;
  const int wave = tid >> 5;
  const int wm   = wave & 3;    // M sub-tile: wm*32
  const int wn   = wave >> 2;   // N sub-tile: wn*64
  const int bm   = blockIdx.y * 128;
  const int bn   = blockIdx.x * 128;
  const int lrow = lane & 15;
  const int kb0  = (lane >> 4) << 3;   // 0 or 8 per documented A layout

  // Per-thread staging coordinates: 4 chunks of 8 f16 for A and for W.
  // chunk = tid + i*256 -> row = chunk>>3 (0..127), kp = (chunk&7)*8.
  // Issue async copies for K-tile kt into stage buffer `buf`.
  auto stage = [&](int kt, int buf) {
    const unsigned base = (unsigned)buf * 65536u;   // bytes per stage = 64KB
#pragma unroll
    for (int i = 0; i < 4; ++i) {
      int chunk = tid + i * 256;
      int row = chunk >> 3;
      int kp  = (chunk & 7) << 3;
      unsigned la = base + (unsigned)(row * 64 + kp) * 2u;            // A tile
      unsigned lw = base + 32768u + (unsigned)(row * 64 + kp) * 2u;   // W tile
      const void* ga = (const void*)(A + (long)(bm + row) * K + kt + kp);
      const void* gw = (const void*)(W + (long)(bn + row) * K + kt + kp);
      // %2 (smem) is unused in the template; it exists only to escape the
      // LDS object so the "memory" clobber covers it.
      asm volatile("global_load_async_to_lds_b128 %0, %1, off"
                   :: "v"(la), "v"(ga), "v"((void*)smem) : "memory");
      asm volatile("global_load_async_to_lds_b128 %0, %1, off"
                   :: "v"(lw), "v"(gw), "v"((void*)smem) : "memory");
    }
  };

  v8f acc[2][4];
#pragma unroll
  for (int mi = 0; mi < 2; ++mi)
#pragma unroll
    for (int ni = 0; ni < 4; ++ni)
#pragma unroll
      for (int e = 0; e < 8; ++e) acc[mi][ni][e] = 0.f;

  stage(0, 0);
  asm volatile("s_wait_asynccnt 0x0" ::: "memory");
  __syncthreads();

  for (int kt = 0; kt < K; kt += 64) {
    const int buf = (kt >> 6) & 1;
    if (kt + 64 < K) stage(kt + 64, buf ^ 1);    // prefetch next tile async

    const _Float16* sA = smem + buf * 32768;     // elements per stage = 32768
    const _Float16* sW = sA + 16384;
#pragma unroll
    for (int kk = 0; kk < 2; ++kk) {
      v16h af[2], bf[4];
#pragma unroll
      for (int mi = 0; mi < 2; ++mi) {
        const _Float16* p = sA + (wm * 32 + mi * 16 + lrow) * 64 + kk * 32 + kb0;
        ((half8*)&af[mi])[0] = *(const half8*)p;
        ((half8*)&af[mi])[1] = *(const half8*)(p + 16);
      }
#pragma unroll
      for (int ni = 0; ni < 4; ++ni) {
        const _Float16* p = sW + (wn * 64 + ni * 16 + lrow) * 64 + kk * 32 + kb0;
        ((half8*)&bf[ni])[0] = *(const half8*)p;
        ((half8*)&bf[ni])[1] = *(const half8*)(p + 16);
      }
#pragma unroll
      for (int mi = 0; mi < 2; ++mi)
#pragma unroll
        for (int ni = 0; ni < 4; ++ni)
          acc[mi][ni] = __builtin_amdgcn_wmma_f32_16x16x32_f16(
              false, af[mi], false, bf[ni], (short)0, acc[mi][ni], false, false);
    }
    // own async copies (next stage) complete + all waves done reading this stage
    asm volatile("s_wait_asynccnt 0x0" ::: "memory");
    __syncthreads();
  }

  const int rbase = (lane >> 4) << 3;   // C/D layout: lanes>=16 hold M=r+8
#pragma unroll
  for (int mi = 0; mi < 2; ++mi)
#pragma unroll
    for (int ni = 0; ni < 4; ++ni)
#pragma unroll
      for (int r = 0; r < 8; ++r) {
        int row = bm + wm * 32 + mi * 16 + rbase + r;
        int col = bn + wn * 64 + ni * 16 + lrow;
        float v = acc[mi][ni][r];
        if (bias)  v += bias[col];
        if (resid) v += resid[(long)row * N + col];
        if (act == 1)      v = fmaxf(v, 0.f);
        else if (act == 2) v = 0.5f * v * (1.f + erff(v * 0.70710678f));
        if (outF) outF[(long)row * N + col] = v;
        if (outH) outH[(long)row * N + col] = (_Float16)v;
      }
}

// ============================================================
// conv2 as implicit-im2col WMMA GEMM.
// M = 19456 rows (b,t,f), K = 4608 (c,kh,kw), N = 512 out channels.
// Output: relu(+bias), stored f16 permuted to [(b,t)][c*19+f] for slin.
// ============================================================
__global__ __launch_bounds__(256)
void conv2_gemm(const _Float16* __restrict__ im,   // [B][512][513][39] f16
                const _Float16* __restrict__ W,    // [512][4608] f16
                const float* __restrict__ bias,
                _Float16* __restrict__ outH) {     // [BT][9728] f16
  __shared__ __align__(16) _Float16 sA[128 * 64];
  __shared__ __align__(16) _Float16 sW[128 * 64];
  const int tid  = threadIdx.x;
  const int lane = tid & 31;
  const int wave = tid >> 5;
  const int wm   = wave & 3;
  const int wn   = wave >> 2;
  const int bm   = blockIdx.y * 128;
  const int bn   = blockIdx.x * 128;
  const int lrow = lane & 15;
  const int kb0  = (lane >> 4) << 3;

  v8f acc[2][4];
#pragma unroll
  for (int mi = 0; mi < 2; ++mi)
#pragma unroll
    for (int ni = 0; ni < 4; ++ni)
#pragma unroll
      for (int e = 0; e < 8; ++e) acc[mi][ni][e] = 0.f;

  const int kp_g   = tid & 63;       // constant per thread
  const int row0_g = tid >> 6;       // 0..3

  for (int kt = 0; kt < K2; kt += 64) {
    // implicit-im2col gather of the 128x64 A tile (32 f16 / thread)
#pragma unroll
    for (int i = 0; i < 32; ++i) {
      int row = row0_g + i * 4;                 // 0..127
      int r   = bm + row;
      int b   = r / (T_ * F2C);
      int rr  = r % (T_ * F2C);
      int t   = rr / F2C;
      int f   = rr % F2C;
      int kg  = kt + kp_g;
      int c   = kg / 9;
      int rem = kg % 9;
      int kh  = rem / 3;
      int kw  = rem % 3;
      sA[row * 64 + kp_g] =
          im[(((long)b * D_ + c) * T1 + (2 * t + kh)) * F1 + (2 * f + kw)];
    }
#pragma unroll
    for (int i = 0; i < 4; ++i) {
      int chunk = tid + i * 256;
      int row = chunk >> 3;
      int kp  = (chunk & 7) << 3;
      *(half8*)(sW + row * 64 + kp) = *(const half8*)(W + (long)(bn + row) * K2 + kt + kp);
    }
    __syncthreads();
#pragma unroll
    for (int kk = 0; kk < 2; ++kk) {
      v16h af[2], bf[4];
#pragma unroll
      for (int mi = 0; mi < 2; ++mi) {
        const _Float16* p = sA + (wm * 32 + mi * 16 + lrow) * 64 + kk * 32 + kb0;
        ((half8*)&af[mi])[0] = *(const half8*)p;
        ((half8*)&af[mi])[1] = *(const half8*)(p + 16);
      }
#pragma unroll
      for (int ni = 0; ni < 4; ++ni) {
        const _Float16* p = sW + (wn * 64 + ni * 16 + lrow) * 64 + kk * 32 + kb0;
        ((half8*)&bf[ni])[0] = *(const half8*)p;
        ((half8*)&bf[ni])[1] = *(const half8*)(p + 16);
      }
#pragma unroll
      for (int mi = 0; mi < 2; ++mi)
#pragma unroll
        for (int ni = 0; ni < 4; ++ni)
          acc[mi][ni] = __builtin_amdgcn_wmma_f32_16x16x32_f16(
              false, af[mi], false, bf[ni], (short)0, acc[mi][ni], false, false);
    }
    __syncthreads();
  }

  const int rbase = (lane >> 4) << 3;
#pragma unroll
  for (int mi = 0; mi < 2; ++mi)
#pragma unroll
    for (int ni = 0; ni < 4; ++ni)
#pragma unroll
      for (int r = 0; r < 8; ++r) {
        int row = bm + wm * 32 + mi * 16 + rbase + r;
        int col = bn + wn * 64 + ni * 16 + lrow;
        int b  = row / (T_ * F2C);
        int rr = row % (T_ * F2C);
        int t  = rr / F2C;
        int f  = rr % F2C;
        float v = fmaxf(acc[mi][ni][r] + bias[col], 0.f);
        outH[((long)(b * T_ + t)) * SLK + col * F2C + f] = (_Float16)v;
      }
}

// ============================================================
// LayerNorm: one 128-thread block per row; optional f32 and f16 outputs.
// Safe in-place (each thread rewrites only elements it read).
// ============================================================
__global__ __launch_bounds__(128)
void ln_kernel(const float* __restrict__ x, const float* __restrict__ g,
               const float* __restrict__ bta, float* __restrict__ of,
               _Float16* __restrict__ oh, int W) {
  __shared__ float s1[128], s2[128];
  const int row = blockIdx.x;
  const float* xp = x + (long)row * W;
  float sum = 0.f, sq = 0.f;
  for (int c = threadIdx.x; c < W; c += 128) {
    float v = xp[c];
    sum += v; sq += v * v;
  }
  s1[threadIdx.x] = sum; s2[threadIdx.x] = sq;
  __syncthreads();
  for (int o = 64; o > 0; o >>= 1) {
    if (threadIdx.x < o) {
      s1[threadIdx.x] += s1[threadIdx.x + o];
      s2[threadIdx.x] += s2[threadIdx.x + o];
    }
    __syncthreads();
  }
  float mean = s1[0] / (float)W;
  float var  = s2[0] / (float)W - mean * mean;
  float rs   = rsqrtf(var + 1e-5f);
  for (int c = threadIdx.x; c < W; c += 128) {
    float v = (xp[c] - mean) * rs * g[c] + bta[c];
    if (of) of[(long)row * W + c] = v;
    if (oh) oh[(long)row * W + c] = (_Float16)v;
  }
}

// ============================================================
// Windowed attention: one thread per (b,h,q). Window <= SEG+LC = 34 keys.
// q: [BT][512] f32, kv: [BT][1024] f32 (k then v), out: [BT][512] f16.
// ============================================================
__global__ __launch_bounds__(256)
void attn_kernel(const float* __restrict__ q, const float* __restrict__ kv,
                 _Float16* __restrict__ out, int out_len) {
  int tid = blockIdx.x * 256 + threadIdx.x;
  if (tid >= B_ * H_ * T_) return;
  int t = tid & (T_ - 1);
  int h = (tid >> 8) & (H_ - 1);
  int b = tid >> 11;
  int seg = (t >> 2) << 2;
  int k0 = seg - LC_; if (k0 < 0) k0 = 0;
  int k1 = seg + SEG_; if (k1 > out_len) k1 = out_len;

  float qv[64];
  const float* qp = q + ((long)(b * T_ + t)) * D_ + h * 64;
#pragma unroll
  for (int d = 0; d < 64; ++d) qv[d] = qp[d] * 0.125f;   // 1/sqrt(64)

  float sc[SEG_ + LC_];
  float mx = -1e30f;
  for (int k = k0; k < k1; ++k) {
    const float* kp = kv + ((long)(b * T_ + k)) * (2 * D_) + h * 64;
    float s = 0.f;
#pragma unroll
    for (int d = 0; d < 64; ++d) s += qv[d] * kp[d];
    sc[k - k0] = s;
    mx = fmaxf(mx, s);
  }
  float den = 0.f;
  float ov[64];
#pragma unroll
  for (int d = 0; d < 64; ++d) ov[d] = 0.f;
  for (int k = k0; k < k1; ++k) {
    float p = __expf(sc[k - k0] - mx);
    den += p;
    const float* vp = kv + ((long)(b * T_ + k)) * (2 * D_) + D_ + h * 64;
#pragma unroll
    for (int d = 0; d < 64; ++d) ov[d] += p * vp[d];
  }
  float inv = 1.f / den;
  _Float16* op = out + ((long)(b * T_ + t)) * D_ + h * 64;
#pragma unroll
  for (int d = 0; d < 64; ++d) op[d] = (_Float16)(ov[d] * inv);
}

// ============================================================
// out_len tail: ((len-1)//2 - 1)//2 for each batch
// ============================================================
__global__ void tail_kernel(const int* __restrict__ lengths, int* __restrict__ out) {
  int b = threadIdx.x;
  if (b < B_) {
    int l = lengths[b];
    out[b] = ((l - 1) / 2 - 1) / 2;
  }
}

// ============================================================
// host orchestration
// ============================================================
static inline void cvt(hipStream_t s, const float* w, _Float16* o, long n) {
  long blocks = (n + 255) / 256;
  if (blocks > (1 << 20)) blocks = (1 << 20);
  f32_to_f16_kernel<<<(unsigned)blocks, 256, 0, s>>>(w, o, n);
}
static inline void gemm(hipStream_t s, const _Float16* A, const _Float16* W,
                        const float* bias, const float* resid,
                        float* oF, _Float16* oH, int M, int N, int K, int act) {
  dim3 g(N / 128, M / 128);
  gemm_wmma<<<g, 256, 0, s>>>(A, W, bias, resid, oF, oH, M, N, K, act);
}

extern "C" void kernel_launch(void* const* d_in, const int* in_sizes, int n_in,
                              void* d_out, int out_size, void* d_ws, size_t ws_size,
                              hipStream_t stream) {
  const float* feats    = (const float*)d_in[0];
  const float* c1w      = (const float*)d_in[1];
  const float* c1b      = (const float*)d_in[2];
  const float* c2w      = (const float*)d_in[3];
  const float* c2b      = (const float*)d_in[4];
  const float* slw      = (const float*)d_in[5];
  const float* slb      = (const float*)d_in[6];
  const float* ln_in_w  = (const float*)d_in[7];
  const float* ln_in_b  = (const float*)d_in[8];
  const float* wq       = (const float*)d_in[9];
  const float* bq       = (const float*)d_in[10];
  const float* wkv      = (const float*)d_in[11];
  const float* bkv      = (const float*)d_in[12];
  const float* wo       = (const float*)d_in[13];
  const float* bo       = (const float*)d_in[14];
  const float* ff_ln_w  = (const float*)d_in[15];
  const float* ff_ln_b  = (const float*)d_in[16];
  const float* ff1w     = (const float*)d_in[17];
  const float* ff1b     = (const float*)d_in[18];
  const float* ff2w     = (const float*)d_in[19];
  const float* ff2b     = (const float*)d_in[20];
  const float* ln_out_w = (const float*)d_in[21];
  const float* ln_out_b = (const float*)d_in[22];
  const float* olw      = (const float*)d_in[23];
  const float* olb      = (const float*)d_in[24];
  const float* fln_w    = (const float*)d_in[25];
  const float* fln_b    = (const float*)d_in[26];
  const int*   lengths  = (const int*)d_in[27];
  (void)in_sizes; (void)n_in; (void)ws_size; (void)out_size;

  // ---- workspace carve-out ----
  char* ws = (char*)d_ws;
  size_t off = 0;
  auto take = [&](size_t bytes) -> char* {
    char* p = ws + off;
    off = (off + bytes + 255) & ~(size_t)255;
    return p;
  };
  _Float16* conv1o  = (_Float16*)take((size_t)B_ * D_ * T1 * F1 * 2);  // 82 MB
  _Float16* slin_in = (_Float16*)take((size_t)BT * SLK * 2);           // 20 MB
  _Float16* wbuf    = (_Float16*)take((size_t)D_ * SLK * 2);           // 10 MB (max weight)
  float*    x       = (float*)   take((size_t)BT * D_ * 4);
  _Float16* xh      = (_Float16*)take((size_t)BT * D_ * 2);
  _Float16* xnh     = (_Float16*)take((size_t)BT * D_ * 2);
  float*    qbuf    = (float*)   take((size_t)BT * D_ * 4);
  float*    kvbuf   = (float*)   take((size_t)BT * 2 * D_ * 4);
  _Float16* attnh   = (_Float16*)take((size_t)BT * D_ * 2);
  _Float16* ff1h    = (_Float16*)take((size_t)BT * FF_ * 2);

  // ---- conv1 (f32 VALU, writes f16) ----
  {
    long total = (long)B_ * D_ * T1 * F1;
    conv1_kernel<<<(unsigned)((total + 255) / 256), 256, 0, stream>>>(feats, c1w, c1b, conv1o);
  }
  // ---- conv2 as implicit-im2col WMMA GEMM ----
  cvt(stream, c2w, wbuf, (long)D_ * K2);
  {
    dim3 g(D_ / 128, M2 / 128);
    conv2_gemm<<<g, 256, 0, stream>>>(conv1o, wbuf, c2b, slin_in);
  }
  // ---- subsample linear: x = slin_in @ slw^T + slb ----
  cvt(stream, slw, wbuf, (long)D_ * SLK);
  gemm(stream, slin_in, wbuf, slb, nullptr, x, nullptr, BT, D_, SLK, 0);

  const int out_len = ((TIN - 1) / 2 - 1) / 2;   // 256 (lengths are all T_IN)

  // ---- 20 transformer layers ----
  for (int i = 0; i < L_; ++i) {
    // xn = LN(x) -> f16
    ln_kernel<<<BT, 128, 0, stream>>>(x, ln_in_w + (long)i * D_, ln_in_b + (long)i * D_,
                                      nullptr, xnh, D_);
    // q = xn @ wq^T + bq
    cvt(stream, wq + (long)i * D_ * D_, wbuf, (long)D_ * D_);
    gemm(stream, xnh, wbuf, bq + (long)i * D_, nullptr, qbuf, nullptr, BT, D_, D_, 0);
    // kv = xn @ wkv^T + bkv
    cvt(stream, wkv + (long)i * 2 * D_ * D_, wbuf, (long)2 * D_ * D_);
    gemm(stream, xnh, wbuf, bkv + (long)i * 2 * D_, nullptr, kvbuf, nullptr, BT, 2 * D_, D_, 0);
    // windowed attention -> f16
    attn_kernel<<<(B_ * H_ * T_) / 256, 256, 0, stream>>>(qbuf, kvbuf, attnh, out_len);
    // x = x + attn @ wo^T + bo   (in-place residual fuse)
    cvt(stream, wo + (long)i * D_ * D_, wbuf, (long)D_ * D_);
    gemm(stream, attnh, wbuf, bo + (long)i * D_, x, x, nullptr, BT, D_, D_, 0);
    // y = LN(x) -> f16
    ln_kernel<<<BT, 128, 0, stream>>>(x, ff_ln_w + (long)i * D_, ff_ln_b + (long)i * D_,
                                      nullptr, xnh, D_);
    // h1 = gelu(y @ ff1w^T + ff1b) -> f16
    cvt(stream, ff1w + (long)i * FF_ * D_, wbuf, (long)FF_ * D_);
    gemm(stream, xnh, wbuf, ff1b + (long)i * FF_, nullptr, nullptr, ff1h, BT, FF_, D_, 2);
    // x = x + h1 @ ff2w^T + ff2b
    cvt(stream, ff2w + (long)i * D_ * FF_, wbuf, (long)D_ * FF_);
    gemm(stream, ff1h, wbuf, ff2b + (long)i * D_, x, x, nullptr, BT, D_, FF_, 0);
    // x = LN(x)  (in place, also emit f16 for the final projection)
    ln_kernel<<<BT, 128, 0, stream>>>(x, ln_out_w + (long)i * D_, ln_out_b + (long)i * D_,
                                      x, xh, D_);
  }

  // ---- final projection + LN, straight into d_out ----
  float* out_f = (float*)d_out;
  cvt(stream, olw, wbuf, (long)OUTW * D_);
  gemm(stream, xh, wbuf, olb, nullptr, out_f, nullptr, BT, OUTW, D_, 0);
  ln_kernel<<<BT, 128, 0, stream>>>(out_f, fln_w, fln_b, out_f, nullptr, OUTW);

  // ---- out_len tail ----
  tail_kernel<<<1, 32, 0, stream>>>(lengths, (int*)(out_f + (long)BT * OUTW));
}
